// GCNII_star_Layer_22127671509147
// MI455X (gfx1250) — matrix operand order repaired and checked by reading, not compiled, not run on hardware
//
#include <hip/hip_runtime.h>
#include <hip/hip_bf16.h>

#define C_DIM 256

typedef _Float16 v8h  __attribute__((ext_vector_type(8)));
typedef _Float16 v16h __attribute__((ext_vector_type(16)));
typedef float    v8f  __attribute__((ext_vector_type(8)));

// Non-returning fp32 atomic add -> guaranteed GLOBAL_ATOMIC_ADD_F32 (STOREcnt-
// tracked; S_ENDPGM performs an implicit wait-idle, so no explicit wait needed).
__device__ __forceinline__ void atomAddF(float* p, float v) {
    asm volatile("global_atomic_add_f32 %0, %1, off" : : "v"(p), "v"(v) : "memory");
}

// ---------------- kernel 0: zero the ax accumulator in workspace ----------------
__global__ void zero_kernel(float4* __restrict__ p, long long n4) {
    long long i = (long long)blockIdx.x * blockDim.x + threadIdx.x;
    if (i < n4) {
        float4 z; z.x = 0.f; z.y = 0.f; z.z = 0.f; z.w = 0.f;
        p[i] = z;
    }
}

// ------- kernel 1: W [256x256] f32 row-major -> Wt [n][k] f16 (transposed) -------
__global__ void wconv_kernel(const float* __restrict__ W, _Float16* __restrict__ Wt) {
    int t = blockIdx.x * blockDim.x + threadIdx.x;  // 65536 threads
    int n = t >> 8;          // output row (original column)
    int k = t & 255;         // output col (original row)
    Wt[n * C_DIM + k] = (_Float16)W[k * C_DIM + n];
}

// ---------------- kernel 2: COO SpMM scatter: ax[row] += val * x[col] ------------
// One wave (32 lanes) per edge; each lane handles 8 contiguous channels.
// Edge metadata is wave-uniform: force scalarization with readfirstlane so
// row/col/val come through SMEM (s_load), not 32 redundant VMEM lanes.
__global__ void spmm_kernel(const float* __restrict__ x,
                            const float* __restrict__ vals,
                            const long long* __restrict__ ei,
                            float* __restrict__ ax, int nE) {
    long long t = (long long)blockIdx.x * blockDim.x + threadIdx.x;
    int e = __builtin_amdgcn_readfirstlane((int)(t >> 5));  // wave-uniform edge id
    if (e >= nE) return;
    int c0 = ((int)t & 31) * 8;

    long long row = ei[e];                 // scalar load
    long long col = ei[(long long)nE + e]; // scalar load
    float v = vals[e];                     // scalar load

    const float4* xs = (const float4*)(x + col * C_DIM + c0);
    float4 a = xs[0];
    float4 b = xs[1];
    float* dst = ax + row * C_DIM + c0;
    atomAddF(dst + 0, v * a.x);
    atomAddF(dst + 1, v * a.y);
    atomAddF(dst + 2, v * a.z);
    atomAddF(dst + 3, v * a.w);
    atomAddF(dst + 4, v * b.x);
    atomAddF(dst + 5, v * b.y);
    atomAddF(dst + 6, v * b.z);
    atomAddF(dst + 7, v * b.w);
}

// ------- kernel 3: fused dual-GEMM + blend epilogue via v_wmma_f32_16x16x32_f16 --
// Block = 256 threads = 8 waves; computes a 16-row x 256-col output strip.
// Wave w owns column tiles [w*32, w*32+16) and [w*32+16, w*32+32).
__global__ void __launch_bounds__(256)
gemm_fused_kernel(const float* __restrict__ ax, const float* __restrict__ h0,
                  const _Float16* __restrict__ W1t, const _Float16* __restrict__ W2t,
                  const float* __restrict__ alpha_p, const float* __restrict__ beta_p,
                  float* __restrict__ out) {
    constexpr int LDA = C_DIM + 8;                // padded half-stride (16B pad)
    __shared__ _Float16 A1[16 * LDA];             // f16 strip of ax
    __shared__ _Float16 A2[16 * LDA];             // f16 strip of h0

    const int t = threadIdx.x;
    const long long r0 = (long long)blockIdx.x * 16;

    // Stage both 16x256 f32 strips into LDS as f16 (each thread: 4 float4 loads).
    for (int i = 0; i < 4; ++i) {
        int idx = t + i * 256;                    // index into 16x64 float4 grid
        int row = idx >> 6;
        int c4  = (idx & 63) * 4;
        float4 va = ((const float4*)(ax + (r0 + row) * C_DIM + c4))[0];
        float4 vh = ((const float4*)(h0 + (r0 + row) * C_DIM + c4))[0];
        _Float16* pa = &A1[row * LDA + c4];
        pa[0] = (_Float16)va.x; pa[1] = (_Float16)va.y;
        pa[2] = (_Float16)va.z; pa[3] = (_Float16)va.w;
        _Float16* ph = &A2[row * LDA + c4];
        ph[0] = (_Float16)vh.x; ph[1] = (_Float16)vh.y;
        ph[2] = (_Float16)vh.z; ph[3] = (_Float16)vh.w;
    }
    __syncthreads();

    const int lane = t & 31;
    const int wave = t >> 5;
    const int m    = lane & 15;       // row within A tile / col within B,C,D tiles
    const int half = lane >> 4;       // lane half selects K sub-range
    const int n0a  = wave * 32;
    const int n0b  = n0a + 16;
    const int colA = n0a + m;         // B-fragment column for tile a
    const int colB = n0b + m;         // B-fragment column for tile b

    v8f acc1a = {}, acc1b = {}, acc2a = {}, acc2b = {};

    for (int kb = 0; kb < 8; ++kb) {
        const int k0 = kb * 32;

        // A fragments (contiguous K to match B): VGPR0-3 <- K k0+half*16..+7,
        // VGPR4-7 <- K k0+half*16+8..+15; two 16-byte LDS reads each.
        v8h a1lo = *(const v8h*)&A1[m * LDA + k0 + half * 16];
        v8h a1hi = *(const v8h*)&A1[m * LDA + k0 + half * 16 + 8];
        v8h a2lo = *(const v8h*)&A2[m * LDA + k0 + half * 16];
        v8h a2hi = *(const v8h*)&A2[m * LDA + k0 + half * 16 + 8];
        v16h a1 = __builtin_shufflevector(a1lo, a1hi, 0,1,2,3,4,5,6,7,8,9,10,11,12,13,14,15);
        v16h a2 = __builtin_shufflevector(a2lo, a2hi, 0,1,2,3,4,5,6,7,8,9,10,11,12,13,14,15);

        // B fragments: lane holds column colX, K = k0 + half*16 .. +15 contiguous
        // in pre-transposed W (32 bytes, 32B-aligned).
        v16h b1a = *(const v16h*)&W1t[colA * C_DIM + k0 + half * 16];
        v16h b1b = *(const v16h*)&W1t[colB * C_DIM + k0 + half * 16];
        v16h b2a = *(const v16h*)&W2t[colA * C_DIM + k0 + half * 16];
        v16h b2b = *(const v16h*)&W2t[colB * C_DIM + k0 + half * 16];

        if (kb < 7) {
            __builtin_prefetch(&W1t[colA * C_DIM + k0 + 32 + half * 16], 0, 3);
            __builtin_prefetch(&W2t[colA * C_DIM + k0 + 32 + half * 16], 0, 3);
        }

        acc1a = __builtin_amdgcn_wmma_f32_16x16x32_f16(false, a1, false, b1a, (short)0, acc1a, false, false);
        acc1b = __builtin_amdgcn_wmma_f32_16x16x32_f16(false, a1, false, b1b, (short)0, acc1b, false, false);
        acc2a = __builtin_amdgcn_wmma_f32_16x16x32_f16(false, a2, false, b2a, (short)0, acc2a, false, false);
        acc2b = __builtin_amdgcn_wmma_f32_16x16x32_f16(false, a2, false, b2b, (short)0, acc2b, false, false);
    }

    // Blend epilogue:
    // out = (1-a)(1-b)*ax + (1-a)b*(ax@W1) + a(1-b)*h0 + ab*(h0@W2)
    const float alpha = alpha_p[0];
    const float beta  = beta_p[0];
    const float c_ax = (1.0f - alpha) * (1.0f - beta);
    const float c_g1 = (1.0f - alpha) * beta;
    const float c_h0 = alpha * (1.0f - beta);
    const float c_g2 = alpha * beta;

    // C/D layout: VGPR r of lane holds (M = half*8 + r, N = lane&15).
    for (int r = 0; r < 8; ++r) {
        long long mrow = r0 + half * 8 + r;
        const float* axr = ax + mrow * C_DIM;
        const float* h0r = h0 + mrow * C_DIM;
        float* outr = out + mrow * C_DIM;
        int na = n0a + m;
        int nb = n0b + m;
        outr[na] = c_ax * axr[na] + c_g1 * acc1a[r] + c_h0 * h0r[na] + c_g2 * acc2a[r];
        outr[nb] = c_ax * axr[nb] + c_g1 * acc1b[r] + c_h0 * h0r[nb] + c_g2 * acc2b[r];
    }
}

extern "C" void kernel_launch(void* const* d_in, const int* in_sizes, int n_in,
                              void* d_out, int out_size, void* d_ws, size_t ws_size,
                              hipStream_t stream) {
    const float*     x    = (const float*)d_in[0];
    const float*     av   = (const float*)d_in[1];
    const long long* ei   = (const long long*)d_in[2];
    const float*     h0   = (const float*)d_in[3];
    const float*     W1   = (const float*)d_in[4];
    const float*     W2   = (const float*)d_in[5];
    const float*     alph = (const float*)d_in[6];
    const float*     beta = (const float*)d_in[7];
    float*           out  = (float*)d_out;

    const int N = in_sizes[0] / C_DIM;   // 100000 (divisible by 16)
    const int E = in_sizes[1];           // 3200000

    // Workspace layout: [ax: N*C f32][W1t: 256*256 f16][W2t: 256*256 f16]
    float* ax = (float*)d_ws;
    size_t axBytes = (size_t)N * C_DIM * sizeof(float);
    _Float16* W1t = (_Float16*)((char*)d_ws + axBytes);
    _Float16* W2t = W1t + C_DIM * C_DIM;

    long long n4 = (long long)N * C_DIM / 4;
    zero_kernel<<<(unsigned)((n4 + 255) / 256), 256, 0, stream>>>((float4*)ax, n4);

    wconv_kernel<<<(C_DIM * C_DIM) / 256, 256, 0, stream>>>(W1, W1t);
    wconv_kernel<<<(C_DIM * C_DIM) / 256, 256, 0, stream>>>(W2, W2t);

    long long spmmThreads = (long long)E * 32;
    spmm_kernel<<<(unsigned)((spmmThreads + 255) / 256), 256, 0, stream>>>(x, av, ei, ax, E);

    gemm_fused_kernel<<<N / 16, 256, 0, stream>>>(ax, h0, W1t, W2t, alph, beta, out);
}
